// TTAggregator_48842368090369
// MI455X (gfx1250) — matrix operand order repaired and checked by reading, not compiled, not run on hardware
//
#include <hip/hip_runtime.h>

#define DEG 10
#define IN 16
#define RANK 16
#define OUT 16

typedef __attribute__((ext_vector_type(16))) __bf16 v16bf;
typedef __attribute__((ext_vector_type(8)))  float  v8f;
typedef __attribute__((ext_vector_type(2)))  __bf16 bf16x2;
typedef __attribute__((ext_vector_type(4)))  unsigned u32x4;
typedef __attribute__((ext_vector_type(8)))  unsigned u32x8;

union PairCast { unsigned u; bf16x2 v; };
union FragCast { unsigned u[8]; v16bf v; };

#define U_BYTES (74 * 1024)          // 72 step-chunks + U0 + U_out, 1 KB each

// ---------- helpers ----------

static __device__ __forceinline__ unsigned pack_bf16x2(float lo, float hi) {
  PairCast t;
  t.v.x = (__bf16)lo;
  t.v.y = (__bf16)hi;
  return t.u;
}

// Load one 16-float row and convert to 8 packed-bf16 dwords (pairs (2v,2v+1)).
static __device__ __forceinline__ void load_row_bf16(const float* __restrict__ p,
                                                     unsigned hp[8]) {
  const float4* v = (const float4*)p;
  float4 x0 = v[0], x1 = v[1], x2 = v[2], x3 = v[3];
  hp[0] = pack_bf16x2(x0.x, x0.y); hp[1] = pack_bf16x2(x0.z, x0.w);
  hp[2] = pack_bf16x2(x1.x, x1.y); hp[3] = pack_bf16x2(x1.z, x1.w);
  hp[4] = pack_bf16x2(x2.x, x2.y); hp[5] = pack_bf16x2(x2.z, x2.w);
  hp[6] = pack_bf16x2(x3.x, x3.y); hp[7] = pack_bf16x2(x3.z, x3.w);
}

// D fragment (lane = column n, VGPR r = row c=r+8*hiGroup) -> per-lane packed
// bf16 ris row: rp[v] = (ris[n,2v], ris[n,2v+1]) valid on ALL 32 lanes.
// Requires one half-wave exchange (lane ^ 16).
static __device__ __forceinline__ void make_ris_packed(const v8f acc,
                                                       unsigned rp[8],
                                                       bool hiGroup) {
  unsigned p[4];
#pragma unroll
  for (int j = 0; j < 4; ++j) p[j] = pack_bf16x2(acc[2 * j], acc[2 * j + 1]);
#pragma unroll
  for (int j = 0; j < 4; ++j) {
    unsigned q = (unsigned)__shfl_xor((int)p[j], 16, 32);
    rp[j]     = hiGroup ? q    : p[j];   // c = 0..7 pairs
    rp[j + 4] = hiGroup ? p[j] : q;      // c = 8..15 pairs
  }
}

// ---------- prep kernel: pack U tensors into bf16 A-fragment layout ----------
// A-fragment (16-bit A 16x32, wave32): lane L holds row M=L&15;
//   lanes 0-15 : elements 0..7 -> K 0..7,  elements 8..15 -> K 16..23
//   lanes 16-31: elements 0..7 -> K 8..15, elements 8..15 -> K 24..31
// Chunks 0..71  : U_rest[d][a][b][c], K = kc*32 + Klocal, a=K>>4, b=K&15, M=c
// Chunk 72      : U0^T   (K=b<16, zero-padded to 32)
// Chunk 73      : U_out^T(K=a<16, zero-padded to 32)
// Storage: w[(chunk*32 + lane)*16 + e]  (1 KB per chunk, 74 KB total)
__global__ void tt_prep(const float* __restrict__ U0,
                        const float* __restrict__ Urest,
                        const float* __restrict__ Uout,
                        __bf16* __restrict__ w) {
  const int chunk = blockIdx.x;        // 0..73
  const int lane  = threadIdx.x;       // 0..31
  const int c  = lane & 15;
  const int hi = lane >> 4;
  __bf16 vals[16];
#pragma unroll
  for (int e = 0; e < 16; ++e) {
    const int Kl = e + 8 * hi + ((e & 8) ? 8 : 0);   // K within 32-chunk
    float v = 0.0f;
    if (chunk < 72) {
      const int d  = chunk >> 3;
      const int kc = chunk & 7;
      const int K  = kc * 32 + Kl;
      const int a  = K >> 4;
      const int b  = K & 15;
      v = Urest[((d * 16 + a) * 16 + b) * 16 + c];
    } else if (chunk == 72) {
      if (Kl < 16) v = U0[Kl * 16 + c];
    } else {
      if (Kl < 16) v = Uout[Kl * 16 + c];
    }
    vals[e] = (__bf16)v;
  }
  __bf16* dst = w + (size_t)(chunk * 32 + lane) * 16;
#pragma unroll
  for (int e = 0; e < 16; ++e) dst[e] = vals[e];
}

// ---------- main kernel: 16 nodes per wave, chained bf16 WMMAs ----------
__global__ __launch_bounds__(256) void tt_agg_main(const float* __restrict__ ns,
                                                   const __bf16* __restrict__ wU,
                                                   float* __restrict__ out,
                                                   int N) {
  __shared__ __align__(16) unsigned char ldsU[U_BYTES];

  // ---- Stage all packed U fragments into LDS with one Tensor Data Mover DMA
  // per workgroup (wave 0 issues; TENSORcnt tracks completion).
  // D# per CDNA5 ISA ch.8: 2-group form (<=2D tensor), data_size=4B,
  // one row of 18944 dwords, no multicast (not in a cluster), no padding.
  {
    const unsigned NDW = U_BYTES / 4;                        // 18944
    const unsigned ldsAddr = (unsigned)(uintptr_t)(&ldsU[0]);
    const unsigned long long ga = (unsigned long long)(uintptr_t)wU;

    u32x4 g0;
    g0[0] = 1u;                                              // count=1, user D#
    g0[1] = ldsAddr;                                         // lds_addr
    g0[2] = (unsigned)(ga & 0xffffffffu);                    // global_addr[31:0]
    g0[3] = (unsigned)((ga >> 32) & 0x01ffffffu)             // global_addr[56:32]
          | 0x80000000u;                                     // type=2 ("image")

    u32x8 g1;
    g1[0] = 0x2u << 16;                                      // data_size=4B
    g1[1] = (NDW & 0xffffu) << 16;                           // tensor_dim0[15:0]
    g1[2] = (NDW >> 16) | (1u << 16);                        // dim0[31:16], tensor_dim1=1
    g1[3] = (NDW & 0xffffu) << 16;                           // tile_dim0 = NDW
    g1[4] = 1u;                                              // tile_dim1=1, tile_dim2=0
    g1[5] = NDW;                                             // tensor_dim0_stride[31:0]
    g1[6] = 0u;                                              // stride hi / dim1_stride lo
    g1[7] = 0u;

    if (threadIdx.x < 32) {                                  // wave 0 only
      asm volatile("tensor_load_to_lds %0, %1"
                   :
                   : "s"(g0), "s"(g1)
                   : "memory");
      __builtin_amdgcn_s_wait_tensorcnt(0);
    }
  }
  __syncthreads();

  const int lane = threadIdx.x & 31;
  const int nl   = lane & 15;
  const bool hiG = lane >= 16;
  const int tiles = (N + 15) >> 4;
  const int waveId     = (int)((blockIdx.x * blockDim.x + threadIdx.x) >> 5);
  const int waveStride = (int)((gridDim.x * blockDim.x) >> 5);

  for (int t = waveId; t < tiles; t += waveStride) {
    const int nodeRaw = t * 16 + nl;
    const int node = (nodeRaw < N) ? nodeRaw : (N - 1);   // clamp: EXEC stays full
    const float* row = ns + (size_t)node * (DEG * IN);

    // hint the next tile's rows toward the caches
    {
      int nn = t + waveStride;
      if (nn < tiles) {
        int pnode = nn * 16 + nl;
        if (pnode >= N) pnode = N - 1;
        __builtin_prefetch(ns + (size_t)pnode * (DEG * IN), 0, 3);
      }
    }

    unsigned hp[8];
    load_row_bf16(row, hp);

    // ris0 = h0 @ U0 : one WMMA, K=16 padded (B elements 8..15 zeroed)
    FragCast b0;
#pragma unroll
    for (int j = 0; j < 8; ++j) b0.u[j] = (j < 4) ? hp[j + (hiG ? 4 : 0)] : 0u;
    v16bf A0 = *(const v16bf*)(ldsU + 72 * 1024 + lane * 32);
    v8f acc = {};
    acc = __builtin_amdgcn_wmma_f32_16x16x32_bf16(false, A0, false, b0.v,
                                                  (short)0, acc, false, false);

    unsigned rp[8];
    for (int d = 1; d < DEG; ++d) {
      make_ris_packed(acc, rp, hiG);
      load_row_bf16(row + d * IN, hp);
      v8f z = {};
      acc = z;
#pragma unroll
      for (int kc = 0; kc < 8; ++kc) {
        v16bf A = *(const v16bf*)(ldsU + ((d - 1) * 8 + kc) * 1024 + lane * 32);
        // splat ris[n, 2*kc + (v >= 4)] into a packed bf16 pair
        const unsigned r = rp[kc];
        // B VGPR v = splat * (h[n,8*hiGroup+2(v&3)], +1)  -> v_pk_mul_bf16
        FragCast bb;
#pragma unroll
        for (int j = 0; j < 8; ++j) {
          PairCast sp;
          sp.u = ((j < 4) ? (r & 0xffffu) : (r >> 16)) * 0x10001u;
          PairCast hh; hh.u = hp[(hiG ? 4 : 0) + (j & 3)];
          PairCast pr; pr.v = sp.v * hh.v;
          bb.u[j] = pr.u;
        }
        acc = __builtin_amdgcn_wmma_f32_16x16x32_bf16(false, A, false, bb.v,
                                                      (short)0, acc, false, false);
      }
    }

    // h_out = ris @ U_output : one WMMA, K=16 padded
    make_ris_packed(acc, rp, hiG);
    FragCast bo;
#pragma unroll
    for (int j = 0; j < 8; ++j) bo.u[j] = (j < 4) ? rp[j + (hiG ? 4 : 0)] : 0u;
    v16bf Ao = *(const v16bf*)(ldsU + 73 * 1024 + lane * 32);
    v8f ho = {};
    ho = __builtin_amdgcn_wmma_f32_16x16x32_bf16(false, Ao, false, bo.v,
                                                 (short)0, ho, false, false);

    // D[o, n]: lane writes 8 consecutive f32 of its node's output row
    if (nodeRaw < N) {
      float* o = out + (size_t)nodeRaw * OUT + (hiG ? 8 : 0);
      float4 s0 = make_float4(ho[0], ho[1], ho[2], ho[3]);
      float4 s1 = make_float4(ho[4], ho[5], ho[6], ho[7]);
      ((float4*)o)[0] = s0;
      ((float4*)o)[1] = s1;
    }
  }
}

// ---------- launch ----------
extern "C" void kernel_launch(void* const* d_in, const int* in_sizes, int n_in,
                              void* d_out, int out_size, void* d_ws, size_t ws_size,
                              hipStream_t stream) {
  const float* ns    = (const float*)d_in[0];   // (N, DEG, IN) f32
  const float* U0    = (const float*)d_in[1];   // (IN, RANK)
  const float* Urest = (const float*)d_in[2];   // (DEG-1, IN, RANK, RANK)
  const float* Uout  = (const float*)d_in[3];   // (RANK, OUT)
  float* out = (float*)d_out;                   // (N, OUT) f32

  const int N = in_sizes[0] / (DEG * IN);
  __bf16* w = (__bf16*)d_ws;                    // needs 74 KB of scratch

  tt_prep<<<74, 32, 0, stream>>>(U0, Urest, Uout, w);

  const int tiles = (N + 15) / 16;
  int blocks = (tiles + 7) / 8;                 // 8 waves per 256-thread block
  if (blocks > 4096) blocks = 4096;
  if (blocks < 1) blocks = 1;
  tt_agg_main<<<blocks, 256, 0, stream>>>(ns, w, out, N);
}
